// RNNDecoder_7267084664849
// MI455X (gfx1250) — matrix operand order repaired and checked by reading, not compiled, not run on hardware
//
#include <hip/hip_runtime.h>

// ---------------------------------------------------------------------------
// Problem constants (match reference)
// ---------------------------------------------------------------------------
#define HD 512
#define VV 32000
#define BB 32
#define SS 64
#define TT 32

typedef __attribute__((ext_vector_type(16))) __bf16 v16bf;
typedef __attribute__((ext_vector_type(8)))  __bf16 v8bf;
typedef __attribute__((ext_vector_type(8)))  float  v8f;

// round-to-nearest-even f32 -> bf16 bits (used only in once-per-launch /
// tiny elementwise kernels, never in the GEMM hot loop)
__device__ __forceinline__ unsigned short f2bfu(float x) {
    union { float f; unsigned u; } v; v.f = x;
    unsigned r = v.u + 0x7FFFu + ((v.u >> 16) & 1u);
    return (unsigned short)(r >> 16);
}

__device__ __forceinline__ v16bf ldfrag(const __bf16* lo, const __bf16* hi) {
    v8bf a = *(const v8bf*)lo;
    v8bf b = *(const v8bf*)hi;
    return __builtin_shufflevector(a, b, 0,1,2,3,4,5,6,7,8,9,10,11,12,13,14,15);
}

#define WMMA_BF16(a, b, c) \
    __builtin_amdgcn_wmma_f32_16x16x32_bf16(false, (a), false, (b), (short)0, (c), false, false)

// ---------------------------------------------------------------------------
// WMMA GEMM:  C[M,N] = A[M,K] * B[N,K]^T + bias[N]
// A, B pre-converted bf16; f32 accumulate; C f32 with row stride ldc.
// One wave computes a 16x64 strip (4 N-tiles), reusing the A fragment.
// Software-pipelined one K-step ahead; sched_group_barrier pins the
// "10 VMEM reads then 4 WMMAs" shape so load waits stagger instead of
// serializing at loadcnt==0.
// Requires M%16==0, N%64==0, K%32==0, K>=64 (true at every call site here).
// ---------------------------------------------------------------------------
__global__ void gemm_bf16wmma(const __bf16* __restrict__ A,
                              const __bf16* __restrict__ Bm,
                              const float* __restrict__ bias,
                              float* __restrict__ C,
                              int M, int N, int K, long ldc) {
    const int lane = threadIdx.x & 31;
    const int wave = threadIdx.x >> 5;
    const int nGroups = N >> 6;                 // 4 tiles of 16 per group
    const int total   = (M >> 4) * nGroups;
    const int grp     = blockIdx.x * 8 + wave;  // wave-uniform
    if (grp >= total) return;                   // whole wave exits together
    const int mt = grp / nGroups;
    const int ng = grp - mt * nGroups;

    const int mrow  = (mt << 4) + (lane & 15);
    const int ncol0 = (ng << 6) + (lane & 15);
    const int kbA   = (lane >= 16) ? 16 : 0;    // A lane-half K base
    const int kbB   = (lane >= 16) ? 16 : 0;    // B lane-half K base

    const __bf16* arow = A + (size_t)mrow * K + kbA;
    const __bf16* br0  = Bm + (size_t)(ncol0 +  0) * K + kbB;
    const __bf16* br1  = Bm + (size_t)(ncol0 + 16) * K + kbB;
    const __bf16* br2  = Bm + (size_t)(ncol0 + 32) * K + kbB;
    const __bf16* br3  = Bm + (size_t)(ncol0 + 48) * K + kbB;

    __builtin_prefetch(br0, 0, 1);              // global_prefetch_b8 (B stream)
    __builtin_prefetch(br2, 0, 1);

    // ---- prologue: fragments for k0 = 0 ----
    v16bf a  = ldfrag(arow, arow + 8);
    v16bf b0 = ldfrag(br0,  br0 + 8);
    v16bf b1 = ldfrag(br1,  br1 + 8);
    v16bf b2 = ldfrag(br2,  br2 + 8);
    v16bf b3 = ldfrag(br3,  br3 + 8);

    v8f c0 = {}, c1 = {}, c2 = {}, c3 = {};
#pragma unroll 2
    for (int k0 = 0; k0 < K - 32; k0 += 32) {
        const int kn = k0 + 32;
        // load next K-step's fragments (10 x b128, distinct registers)
        v16bf an  = ldfrag(arow + kn, arow + kn + 8);
        v16bf nb0 = ldfrag(br0 + kn,  br0 + kn + 8);
        v16bf nb1 = ldfrag(br1 + kn,  br1 + kn + 8);
        v16bf nb2 = ldfrag(br2 + kn,  br2 + kn + 8);
        v16bf nb3 = ldfrag(br3 + kn,  br3 + kn + 8);
        // compute on current fragments (loaded one K-step earlier)
        c0 = WMMA_BF16(a, b0, c0);
        c1 = WMMA_BF16(a, b1, c1);
        c2 = WMMA_BF16(a, b2, c2);
        c3 = WMMA_BF16(a, b3, c3);
        // pin schedule: all 10 VMEM reads, then the 4 WMMAs
        __builtin_amdgcn_sched_group_barrier(0x020, 10, 0); // VMEM read
        __builtin_amdgcn_sched_group_barrier(0x008, 4, 0);  // MFMA/WMMA
        a = an; b0 = nb0; b1 = nb1; b2 = nb2; b3 = nb3;
    }
    // ---- epilogue: last K-step ----
    c0 = WMMA_BF16(a, b0, c0);
    c1 = WMMA_BF16(a, b1, c1);
    c2 = WMMA_BF16(a, b2, c2);
    c3 = WMMA_BF16(a, b3, c3);

    // C layout: VGPR r, lanes 0-15 -> M=r, lanes 16-31 -> M=r+8; N = lane&15
    const int rbase = (mt << 4) + ((lane >= 16) ? 8 : 0);
#pragma unroll
    for (int nt = 0; nt < 4; ++nt) {
        const int ncol = ncol0 + nt * 16;
        const float bval = bias[ncol];
        const v8f& c = (nt == 0) ? c0 : (nt == 1) ? c1 : (nt == 2) ? c2 : c3;
#pragma unroll
        for (int r = 0; r < 8; ++r)
            C[(size_t)(rbase + r) * ldc + ncol] = c[r] + bval;
    }
}

// ---------------------------------------------------------------------------
// One-time f32 -> bf16 conversion (one element per thread, runs outside the T loop)
// ---------------------------------------------------------------------------
__global__ void f32_to_bf16(const float* __restrict__ src,
                            unsigned short* __restrict__ dst, int n) {
    const int i = blockIdx.x * blockDim.x + threadIdx.x;
    if (i < n) dst[i] = f2bfu(src[i]);
}

// ---------------------------------------------------------------------------
// Attention step (block per batch element, 64 threads = one per s):
//   scores -> softmax -> attn out, context, x = [emb[tok], ctx] in bf16
// ---------------------------------------------------------------------------
__global__ void attn_step(const float* __restrict__ q,
                          const float* __restrict__ ua,
                          const float* __restrict__ Va_w,
                          const float* __restrict__ Va_b,
                          const float* __restrict__ enc,
                          const float* __restrict__ emb,
                          const int* __restrict__ target,
                          unsigned short* __restrict__ xbf,
                          float* __restrict__ attn_out,
                          int t) {
    __shared__ float qs[HD];
    __shared__ float wsh[SS];
    __shared__ float red[SS];
    const int b = blockIdx.x;
    const int s = threadIdx.x;

    for (int i = s; i < HD; i += SS) qs[i] = q[b * HD + i];
    __syncthreads();

    const float* uar = ua + ((size_t)(b * SS + s)) * HD;
    float acc = 0.f;
    for (int h = 0; h < HD; ++h) acc += tanhf(qs[h] + uar[h]) * Va_w[h];
    acc += Va_b[0];

    // softmax over S=64
    red[s] = acc; __syncthreads();
    for (int k = 32; k > 0; k >>= 1) {
        if (s < k) red[s] = fmaxf(red[s], red[s + k]);
        __syncthreads();
    }
    const float mx = red[0]; __syncthreads();
    const float e = expf(acc - mx);
    red[s] = e; __syncthreads();
    for (int k = 32; k > 0; k >>= 1) {
        if (s < k) red[s] += red[s + k];
        __syncthreads();
    }
    const float w = e / red[0];
    wsh[s] = w;
    attn_out[((size_t)b * TT + t) * SS + s] = w;
    __syncthreads();

    // ctx[k] = sum_s w[s] * enc[b,s,k]  -> x[H + k] (bf16)
    for (int k = s; k < 2 * HD; k += SS) {
        float a2 = 0.f;
        for (int ss2 = 0; ss2 < SS; ++ss2)
            a2 += wsh[ss2] * enc[((size_t)(b * SS + ss2)) * (2 * HD) + k];
        xbf[b * 3 * HD + HD + k] = f2bfu(a2);
    }
    // x[0:H] = emb[tok] (bf16); teacher forcing, t==0 -> SOS token 0
    const int tok = (t == 0) ? 0 : target[b * TT + (t - 1)];
    for (int i = s; i < HD; i += SS)
        xbf[b * 3 * HD + i] = f2bfu(emb[(size_t)tok * HD + i]);
}

// ---------------------------------------------------------------------------
// GRU gates: h = (1-z)*n + z*h  (writes f32 master + bf16 mirror)
// ---------------------------------------------------------------------------
__global__ void gru_step(const float* __restrict__ gx,
                         const float* __restrict__ gh,
                         float* __restrict__ h,
                         unsigned short* __restrict__ hbf) {
    const int idx = blockIdx.x * blockDim.x + threadIdx.x;   // BB*HD threads
    const int b = idx >> 9;
    const int j = idx & (HD - 1);
    const int base = b * 3 * HD;
    const float r = 1.f / (1.f + expf(-(gx[base + j] + gh[base + j])));
    const float z = 1.f / (1.f + expf(-(gx[base + HD + j] + gh[base + HD + j])));
    const float n = tanhf(gx[base + 2 * HD + j] + r * gh[base + 2 * HD + j]);
    const float hn = (1.f - z) * n + z * h[idx];
    h[idx] = hn;
    hbf[idx] = f2bfu(hn);
}

// ---------------------------------------------------------------------------
// In-place row-wise log-softmax (one block per row)
// ---------------------------------------------------------------------------
__global__ void log_softmax_rows(float* __restrict__ base, long rowStride, int n) {
    __shared__ float red[256];
    float* p = base + (size_t)blockIdx.x * rowStride;
    const int tid = threadIdx.x;
    float m = -3.4e38f;
    for (int i = tid; i < n; i += 256) m = fmaxf(m, p[i]);
    red[tid] = m; __syncthreads();
    for (int k = 128; k > 0; k >>= 1) {
        if (tid < k) red[tid] = fmaxf(red[tid], red[tid + k]);
        __syncthreads();
    }
    m = red[0]; __syncthreads();
    float s = 0.f;
    for (int i = tid; i < n; i += 256) s += expf(p[i] - m);
    red[tid] = s; __syncthreads();
    for (int k = 128; k > 0; k >>= 1) {
        if (tid < k) red[tid] += red[tid + k];
        __syncthreads();
    }
    const float lse = m + logf(red[0]);
    __syncthreads();
    for (int i = tid; i < n; i += 256) p[i] = p[i] - lse;
}

__global__ void copy_f32(const float* __restrict__ src, float* __restrict__ dst, int n) {
    const int i = blockIdx.x * blockDim.x + threadIdx.x;
    if (i < n) dst[i] = src[i];
}

// ---------------------------------------------------------------------------
// Host driver
// ---------------------------------------------------------------------------
extern "C" void kernel_launch(void* const* d_in, const int* in_sizes, int n_in,
                              void* d_out, int out_size, void* d_ws, size_t ws_size,
                              hipStream_t stream) {
    const float* enc_out    = (const float*)d_in[0];   // [B,S,2H]
    const float* enc_hidden = (const float*)d_in[1];   // [1,B,2H]
    const int*   target     = (const int*)  d_in[2];   // [B,T]
    const float* emb        = (const float*)d_in[3];   // [V,H]
    const float* Wa_w       = (const float*)d_in[4];   // [H,H]
    const float* Wa_b       = (const float*)d_in[5];   // [H]
    const float* Ua_w       = (const float*)d_in[6];   // [H,2H]
    const float* Ua_b       = (const float*)d_in[7];   // [H]
    const float* Va_w       = (const float*)d_in[8];   // [1,H]
    const float* Va_b       = (const float*)d_in[9];   // [1]
    const float* W_ih       = (const float*)d_in[10];  // [3H,3H]
    const float* b_ih       = (const float*)d_in[11];  // [3H]
    const float* W_hh       = (const float*)d_in[12];  // [3H,H]
    const float* b_hh       = (const float*)d_in[13];  // [3H]
    const float* Wh         = (const float*)d_in[14];  // [H,2H]
    const float* bh         = (const float*)d_in[15];  // [H]
    const float* out_w      = (const float*)d_in[16];  // [V,H]
    const float* out_b      = (const float*)d_in[17];  // [V]

    // d_out layout: log_probs [B,T,V] ++ h_final [1,B,H] ++ attn [B,T,S]
    float* logp = (float*)d_out;
    float* hfin = logp + (size_t)BB * TT * VV;
    float* attn = hfin + (size_t)BB * HD;

    // ---- workspace carve-up (bf16 mirrors first, then f32 buffers) ----
    char* cur = (char*)d_ws;
    auto takeBf = [&](size_t nelem) {
        unsigned short* p = (unsigned short*)cur; cur += nelem * 2; return p;
    };
    unsigned short* outw_bf = takeBf((size_t)VV * HD);
    unsigned short* wih_bf  = takeBf((size_t)3 * HD * 3 * HD);
    unsigned short* whh_bf  = takeBf((size_t)3 * HD * HD);
    unsigned short* waw_bf  = takeBf((size_t)HD * HD);
    unsigned short* uaw_bf  = takeBf((size_t)HD * 2 * HD);
    unsigned short* wh_bf   = takeBf((size_t)HD * 2 * HD);
    unsigned short* enc_bf  = takeBf((size_t)BB * SS * 2 * HD);
    unsigned short* ench_bf = takeBf((size_t)BB * 2 * HD);
    unsigned short* hbf     = takeBf((size_t)BB * HD);
    unsigned short* xbf     = takeBf((size_t)BB * 3 * HD);
    auto takeF = [&](size_t nelem) {
        float* p = (float*)cur; cur += nelem * 4; return p;
    };
    float* ua_keys = takeF((size_t)BB * SS * HD);
    float* hbuf    = takeF((size_t)BB * HD);
    float* qbuf    = takeF((size_t)BB * HD);
    float* gxbuf   = takeF((size_t)BB * 3 * HD);
    float* ghbuf   = takeF((size_t)BB * 3 * HD);

    auto conv = [&](const float* s, unsigned short* d, int n) {
        f32_to_bf16<<<(n + 255) / 256, 256, 0, stream>>>(s, d, n);
    };
    auto gemm = [&](const unsigned short* A, const unsigned short* Bm,
                    const float* bias, float* C, int M, int N, int K, long ldc) {
        const int waves  = (M / 16) * (N / 64);
        const int blocks = (waves + 7) / 8;
        gemm_bf16wmma<<<blocks, 256, 0, stream>>>(
            (const __bf16*)A, (const __bf16*)Bm, bias, C, M, N, K, ldc);
    };

    // ---- one-time weight / input conversion to bf16 ----
    conv(out_w,      outw_bf, VV * HD);
    conv(W_ih,       wih_bf,  3 * HD * 3 * HD);
    conv(W_hh,       whh_bf,  3 * HD * HD);
    conv(Wa_w,       waw_bf,  HD * HD);
    conv(Ua_w,       uaw_bf,  HD * 2 * HD);
    conv(Wh,         wh_bf,   HD * 2 * HD);
    conv(enc_out,    enc_bf,  BB * SS * 2 * HD);
    conv(enc_hidden, ench_bf, BB * 2 * HD);

    // ua_keys = enc_out @ Ua_w^T + Ua_b   [B*S, H]
    gemm(enc_bf, uaw_bf, Ua_b, ua_keys, BB * SS, HD, 2 * HD, HD);
    // h0 = enc_hidden @ Wh^T + bh
    gemm(ench_bf, wh_bf, bh, hbuf, BB, HD, 2 * HD, HD);
    conv(hbuf, hbf, BB * HD);

    for (int t = 0; t < TT; ++t) {
        // q = h @ Wa^T + Wa_b
        gemm(hbf, waw_bf, Wa_b, qbuf, BB, HD, HD, HD);
        // attention; builds x = [emb[tok], ctx] in bf16
        attn_step<<<BB, SS, 0, stream>>>(qbuf, ua_keys, Va_w, Va_b, enc_out,
                                         emb, target, xbf, attn, t);
        // gx = x @ W_ih^T + b_ih ; gh = h @ W_hh^T + b_hh
        gemm(xbf, wih_bf, b_ih, gxbuf, BB, 3 * HD, 3 * HD, 3 * HD);
        gemm(hbf, whh_bf, b_hh, ghbuf, BB, 3 * HD, HD, 3 * HD);
        // GRU update (f32 master + bf16 mirror)
        gru_step<<<(BB * HD) / 256, 256, 0, stream>>>(gxbuf, ghbuf, hbuf, hbf);
        // logits straight into d_out: row stride T*V, row base t*V
        gemm(hbf, outw_bf, out_b, logp + (size_t)t * VV, BB, VV, HD, (long)TT * VV);
        // in-place log-softmax of this step's 32 rows
        log_softmax_rows<<<BB, 256, 0, stream>>>(logp + (size_t)t * VV,
                                                 (long)TT * VV, VV);
    }

    // h_final
    copy_f32<<<(BB * HD + 255) / 256, 256, 0, stream>>>(hbuf, hfin, BB * HD);
}
